// GPTLanguageModel_90005334655530
// MI455X (gfx1250) — matrix operand order repaired and checked by reading, not compiled, not run on hardware
//
#include <hip/hip_runtime.h>
#include <hip/hip_bf16.h>

// ---------------- model constants ----------------
constexpr int Bc = 4, Tc = 1024, Dc = 1024, Hc = 16, Lc = 12;
constexpr int FFc = 4 * Dc, Vc = 32000, HDc = Dc / Hc;       // 4096, 32000, 64
constexpr int BT = Bc * Tc;                                   // 4096 rows
constexpr int QKVN = 3 * Dc;                                  // 3072

// ---------------- vector types / WMMA helpers ----------------
typedef __attribute__((ext_vector_type(16))) __bf16 v16bf;
typedef __attribute__((ext_vector_type(8)))  __bf16 v8bf;
typedef __attribute__((ext_vector_type(8)))  float  v8f;
typedef __attribute__((ext_vector_type(8)))  unsigned int v8u;

__device__ __forceinline__ v8f wmma_bf16(v16bf a, v16bf b, v8f c) {
  return __builtin_amdgcn_wmma_f32_16x16x32_bf16(
      /*neg_a=*/false, a, /*neg_b=*/false, b,
      /*c_mod=*/(short)0, c, /*reuse_a=*/false, /*reuse_b=*/false);
}

// A-fragment (16x32, 16-bit): lanes 0-15 hold K=[0..7]+[16..23], lanes 16-31 K=[8..15]+[24..31]
__device__ __forceinline__ v16bf ld_afrag(const __bf16* base, int ld, int kbase) {
  const int lane = threadIdx.x & 31;
  const int row  = lane & 15;
  const int c    = kbase + ((lane >> 4) << 3);
  const __bf16* p = base + row * ld + c;
  v8bf lo = *(const v8bf*)p;
  v8bf hi = *(const v8bf*)(p + 16);
  v16bf r;
#pragma unroll
  for (int i = 0; i < 8; ++i) { r[i] = lo[i]; r[i + 8] = hi[i]; }
  return r;
}

// B-fragment (32x16, 16-bit), operand stored as B^T rows ([N,K]):
// lane n=lane%16 holds K=[0..7]+[16..23] (lanes 0-15), K=[8..15]+[24..31] (lanes 16-31).
__device__ __forceinline__ v16bf ld_bfrag(const __bf16* base, int ld, int kbase) {
  const int lane = threadIdx.x & 31;
  const int row  = lane & 15;
  const int c    = kbase + ((lane >> 4) << 3);
  const __bf16* p = base + row * ld + c;
  v8bf lo = *(const v8bf*)p;
  v8bf hi = *(const v8bf*)(p + 16);
  v16bf r;
#pragma unroll
  for (int i = 0; i < 8; ++i) { r[i] = lo[i]; r[i + 8] = hi[i]; }
  return r;
}

// async global->LDS copy of 16 bytes per lane (tracked by ASYNCcnt)
__device__ __forceinline__ void async_cp16(unsigned lds_addr, const __bf16* gp) {
  asm volatile("global_load_async_to_lds_b128 %0, %1, off"
               :: "v"(lds_addr), "v"((unsigned long long)gp) : "memory");
}
__device__ __forceinline__ unsigned lds_lo32(const void* p) {
  return (unsigned)(unsigned long long)p;   // generic LDS addr: low 32 bits = group-segment offset
}

// XOR butterfly via immediate ds_swizzle (and=0x1f, or=0, xor=m)
#define SWZ_XOR_F(v, m) \
  __uint_as_float((unsigned)__builtin_amdgcn_ds_swizzle((int)__float_as_uint(v), (0x1f | ((m) << 10))))

// ---------------- embedding ----------------
__global__ __launch_bounds__(256) void embed_k(const int* __restrict__ idx,
                                               const float* __restrict__ tok,
                                               const float* __restrict__ pos,
                                               float* __restrict__ x) {
  const int row = blockIdx.x;            // b*T + t
  const int tid = idx[row];
  const int t   = row & (Tc - 1);
  const float* ts = tok + (size_t)tid * Dc;
  const float* ps = pos + (size_t)t * Dc;
  float* xr = x + (size_t)row * Dc;
  for (int i = threadIdx.x; i < Dc; i += 256) xr[i] = ts[i] + ps[i];
}

// ---------------- layernorm -> bf16 ----------------
__global__ __launch_bounds__(256) void ln_k(const float* __restrict__ x,
                                            const float* __restrict__ g,
                                            const float* __restrict__ bb,
                                            __bf16* __restrict__ out) {
  __shared__ float red[256];
  const int row = blockIdx.x, t = threadIdx.x;
  const float* xr = x + (size_t)row * Dc;
  float v[4], s = 0.f, s2 = 0.f;
#pragma unroll
  for (int i = 0; i < 4; ++i) {
    v[i] = xr[t + i * 256];
    s += v[i]; s2 += v[i] * v[i];
  }
  red[t] = s; __syncthreads();
  for (int st = 128; st; st >>= 1) { if (t < st) red[t] += red[t + st]; __syncthreads(); }
  const float mean = red[0] * (1.f / Dc);
  __syncthreads();
  red[t] = s2; __syncthreads();
  for (int st = 128; st; st >>= 1) { if (t < st) red[t] += red[t + st]; __syncthreads(); }
  const float var = red[0] * (1.f / Dc) - mean * mean;
  const float rs = rsqrtf(var + 1e-5f);
  __bf16* orow = out + (size_t)row * Dc;
#pragma unroll
  for (int i = 0; i < 4; ++i) {
    const int c = t + i * 256;
    orow[c] = (__bf16)((v[i] - mean) * rs * g[c] + bb[c]);
  }
}

// ---------------- weight convert f32 -> bf16 transposed ([K,N] -> [N,K]) ----------------
__global__ __launch_bounds__(256) void convT_k(const float* __restrict__ w,
                                               __bf16* __restrict__ wt,
                                               int K, int N) {
  const size_t tid = (size_t)blockIdx.x * 256 + threadIdx.x;
  const size_t n = tid / (size_t)K, k = tid % (size_t)K;
  wt[tid] = (__bf16)w[k * (size_t)N + n];
}

// wqkv layer slice [3,H,D,HD] -> wt[n*D + d], n = which*1024 + h*64 + hd
__global__ __launch_bounds__(256) void convqkv_k(const float* __restrict__ w,
                                                 __bf16* __restrict__ wt) {
  const int tid = blockIdx.x * 256 + threadIdx.x;   // 3072*1024 total
  const int n = tid >> 10, d = tid & 1023;
  const int which = n >> 10, rem = n & 1023, h = rem >> 6, hd = rem & 63;
  wt[tid] = (__bf16)w[((size_t)(which * Hc + h) * Dc + d) * HDc + hd];
}

// ---------------- generic bf16 GEMM: C[M,N] = A[M,K] * Bt^T + epilogue ----------------
// 128x128 tile, 256 threads (8 waves as 2x4), K-stage 64, double-buffered LDS filled
// by global_load_async_to_lds_b128 (ASYNCcnt-pipelined).
// MODE 0: bf16 out = acc             (QKV)
// MODE 1: bf16 out = gelu(acc+bias)  (FFN up)
// MODE 2: f32  out = resid+acc+bias  (proj / FFN down, in-place residual)
// MODE 3: f32  out = acc+bias        (LM head)
constexpr int GST = 72;   // padded LDS row stride (elements): 144B, 16B-aligned, bank-friendly
template <int MODE>
__global__ __launch_bounds__(256) void gemm_bf16_k(const __bf16* __restrict__ A, int lda,
                                                   const __bf16* __restrict__ Bt, int ldb,
                                                   const float* __restrict__ bias,
                                                   const float* __restrict__ resid,
                                                   float* __restrict__ outf,
                                                   __bf16* __restrict__ outb,
                                                   int ldc, int K) {
  __shared__ alignas(32) __bf16 lA[2][128 * GST];
  __shared__ alignas(32) __bf16 lB[2][128 * GST];
  const int m0 = blockIdx.y * 128, n0 = blockIdx.x * 128;
  const int t = threadIdx.x, w = t >> 5, lane = t & 31;
  const int wm = (w >> 2) * 64, wn = (w & 3) * 32;     // wave sub-tile: 64 rows x 32 cols
  const int lrow = t >> 1, lseg = (t & 1) * 32;        // 2 threads per 64-elem row

  const __bf16* gA = A + (size_t)(m0 + lrow) * lda + lseg;
  const __bf16* gB = Bt + (size_t)(n0 + lrow) * ldb + lseg;
  const unsigned laA = lds_lo32(&lA[0][lrow * GST + lseg]);
  const unsigned laB = lds_lo32(&lB[0][lrow * GST + lseg]);
  const unsigned bufB = (unsigned)(128 * GST * 2);     // bytes per buffer

  auto issue = [&](int s) {
    const __bf16* pa = gA + (size_t)s * 64;
    const __bf16* pb = gB + (size_t)s * 64;
    const unsigned la = laA + (unsigned)(s & 1) * bufB;
    const unsigned lb = laB + (unsigned)(s & 1) * bufB;
#pragma unroll
    for (int q = 0; q < 4; ++q) {
      async_cp16(la + q * 16, pa + q * 8);
      async_cp16(lb + q * 16, pb + q * 8);
    }
  };

  v8f acc[4][2] = {};
  const int S = K >> 6;
  issue(0);
  for (int s = 0; s < S; ++s) {
    if (s + 1 < S) {
      issue(s + 1);                                      // prefetch next stage
      asm volatile("s_wait_asynccnt 0x8" ::: "memory");  // older 8 (stage s) have landed
    } else {
      asm volatile("s_wait_asynccnt 0x0" ::: "memory");
    }
    __syncthreads();
    const __bf16* tA = lA[s & 1];
    const __bf16* tB = lB[s & 1];
#pragma unroll
    for (int kk = 0; kk < 64; kk += 32) {
      v16bf af[4], bfr[2];
#pragma unroll
      for (int i = 0; i < 4; ++i) af[i] = ld_afrag(tA + (wm + i * 16) * GST, GST, kk);
#pragma unroll
      for (int j = 0; j < 2; ++j) bfr[j] = ld_bfrag(tB + (wn + j * 16) * GST, GST, kk);
#pragma unroll
      for (int i = 0; i < 4; ++i)
#pragma unroll
        for (int j = 0; j < 2; ++j)
          acc[i][j] = wmma_bf16(af[i], bfr[j], acc[i][j]);
    }
    __syncthreads();   // all waves done reading buf (s&1) before stage s+2 overwrites it
  }

  const int rbase = (lane >> 4) << 3;
#pragma unroll
  for (int i = 0; i < 4; ++i)
#pragma unroll
    for (int j = 0; j < 2; ++j) {
      const int cm = m0 + wm + i * 16;
      const int cn = n0 + wn + j * 16 + (lane & 15);
      const float bv = bias ? bias[cn] : 0.0f;
#pragma unroll
      for (int r = 0; r < 8; ++r) {
        const int row = cm + rbase + r;
        const size_t o = (size_t)row * ldc + cn;
        const float v = acc[i][j][r] + bv;
        if (MODE == 0)      outb[o] = (__bf16)v;
        else if (MODE == 1) outb[o] = (__bf16)(0.5f * v * (1.0f + erff(v * 0.70710678118f)));
        else if (MODE == 2) outf[o] = resid[o] + v;
        else                outf[o] = v;
      }
    }
}

// ---------------- flash attention (causal), HD=64, key tiles of 64 ----------------
// qkv: [B*T, 3072] bf16 (k|q|v per head).  attn: [B*T, 1024] bf16 (head-concat).
constexpr int AST = 72;   // padded LDS stride for attention tiles
__global__ __launch_bounds__(128) void attn_k(const __bf16* __restrict__ qkv,
                                              __bf16* __restrict__ attn) {
  __shared__ alignas(32) __bf16 lQ[64 * AST];
  __shared__ alignas(32) __bf16 lK[64 * AST];
  __shared__ alignas(32) __bf16 lVT[64 * AST];
  __shared__ alignas(32) __bf16 lP[4 * 16 * AST];

  const int bh = blockIdx.y, b = bh >> 4, h = bh & 15;
  const int q0 = blockIdx.x * 64;
  const size_t rb = (size_t)b * Tc * QKVN;
  const __bf16* Kg = qkv + rb + 0 * Dc + h * HDc;
  const __bf16* Qg = qkv + rb + 1 * Dc + h * HDc;
  const __bf16* Vg = qkv + rb + 2 * Dc + h * HDc;

  const int t = threadIdx.x, w = t >> 5, lane = t & 31;
  const int qw = w * 16;

  { // load Q tile (64 rows x 64)
    const int r = t >> 1, c = (t & 1) * 32;
    *(v8u*)&lQ[r * AST + c]      = *(const v8u*)&Qg[(size_t)(q0 + r) * QKVN + c];
    *(v8u*)&lQ[r * AST + c + 16] = *(const v8u*)&Qg[(size_t)(q0 + r) * QKVN + c + 16];
  }

  v8f o[4] = {};
  float mrun[8], lrun[8];
#pragma unroll
  for (int r = 0; r < 8; ++r) { mrun[r] = -1e30f; lrun[r] = 0.f; }
  const float scale = 0.03125f;   // D**-0.5 (reference scales by n_embd, not head_size)

  for (int jt = 0; jt <= (int)blockIdx.x; ++jt) {
    __syncthreads();
    { // load K tile row-major + V tile transposed
      const int r = t >> 1, c = (t & 1) * 32;
      *(v8u*)&lK[r * AST + c]      = *(const v8u*)&Kg[(size_t)(jt * 64 + r) * QKVN + c];
      *(v8u*)&lK[r * AST + c + 16] = *(const v8u*)&Kg[(size_t)(jt * 64 + r) * QKVN + c + 16];
      __bf16 tmp[32];
      *(v8u*)&tmp[0]  = *(const v8u*)&Vg[(size_t)(jt * 64 + r) * QKVN + c];
      *(v8u*)&tmp[16] = *(const v8u*)&Vg[(size_t)(jt * 64 + r) * QKVN + c + 16];
#pragma unroll
      for (int i = 0; i < 32; ++i) lVT[(c + i) * AST + r] = tmp[i];
    }
    __syncthreads();

    // S = Q * K^T  (16 queries x 64 keys per wave)
    v8f s[4] = {};
#pragma unroll
    for (int ks = 0; ks < 2; ++ks) {
      const v16bf aq = ld_afrag(lQ + qw * AST, AST, ks * 32);
#pragma unroll
      for (int a = 0; a < 4; ++a) {
        const v16bf bk = ld_bfrag(lK + a * 16 * AST, AST, ks * 32);
        s[a] = wmma_bf16(aq, bk, s[a]);
      }
    }

    // online softmax
    const int keyc = jt * 64 + (lane & 15);
    const int qr   = q0 + qw + ((lane >> 4) << 3);
#pragma unroll
    for (int r = 0; r < 8; ++r) {
      float pv[4], mt = -1e30f;
#pragma unroll
      for (int a = 0; a < 4; ++a) {
        float v = s[a][r] * scale;
        if (keyc + a * 16 > qr + r) v = -1e30f;   // causal mask
        pv[a] = v; mt = fmaxf(mt, v);
      }
      mt = fmaxf(mt, SWZ_XOR_F(mt, 8));
      mt = fmaxf(mt, SWZ_XOR_F(mt, 4));
      mt = fmaxf(mt, SWZ_XOR_F(mt, 2));
      mt = fmaxf(mt, SWZ_XOR_F(mt, 1));
      const float mnew = fmaxf(mrun[r], mt);
      const float corr = __expf(mrun[r] - mnew);
      mrun[r] = mnew;
      float ls = 0.f;
#pragma unroll
      for (int a = 0; a < 4; ++a) { const float p = __expf(pv[a] - mnew); s[a][r] = p; ls += p; }
      ls += SWZ_XOR_F(ls, 8);
      ls += SWZ_XOR_F(ls, 4);
      ls += SWZ_XOR_F(ls, 2);
      ls += SWZ_XOR_F(ls, 1);
      lrun[r] = lrun[r] * corr + ls;
#pragma unroll
      for (int a = 0; a < 4; ++a) o[a][r] = o[a][r] * corr;
    }

    // transpose P through per-wave LDS to A-fragment layout
    __bf16* Pw = lP + w * 16 * AST;
#pragma unroll
    for (int a = 0; a < 4; ++a)
#pragma unroll
      for (int r = 0; r < 8; ++r)
        Pw[(((lane >> 4) << 3) + r) * AST + a * 16 + (lane & 15)] = (__bf16)s[a][r];
    asm volatile("" ::: "memory");   // keep DS store->load order (LDS is in-order per wave)

    // O += P * V
#pragma unroll
    for (int ks = 0; ks < 2; ++ks) {
      const v16bf ap = ld_afrag(Pw, AST, ks * 32);
#pragma unroll
      for (int a = 0; a < 4; ++a) {
        const v16bf bv = ld_bfrag(lVT + a * 16 * AST, AST, ks * 32);
        o[a] = wmma_bf16(ap, bv, o[a]);
      }
    }
  }

  // write normalized output, head-concat layout
  const int qrow = q0 + qw + ((lane >> 4) << 3);
#pragma unroll
  for (int a = 0; a < 4; ++a)
#pragma unroll
    for (int r = 0; r < 8; ++r) {
      const float v = o[a][r] / lrun[r];
      attn[(size_t)(b * Tc + qrow + r) * Dc + h * HDc + a * 16 + (lane & 15)] = (__bf16)v;
    }
}

// ---------------- host launcher ----------------
extern "C" void kernel_launch(void* const* d_in, const int* in_sizes, int n_in,
                              void* d_out, int out_size, void* d_ws, size_t ws_size,
                              hipStream_t stream) {
  (void)in_sizes; (void)n_in; (void)out_size; (void)ws_size;
  const int*   idx  = (const int*)d_in[0];
  const float* tok  = (const float*)d_in[1];
  const float* pos  = (const float*)d_in[2];
  const float* wqkv = (const float*)d_in[3];
  const float* wo   = (const float*)d_in[4];
  const float* bo   = (const float*)d_in[5];
  const float* ln1g = (const float*)d_in[6];
  const float* ln1b = (const float*)d_in[7];
  const float* ln2g = (const float*)d_in[8];
  const float* ln2b = (const float*)d_in[9];
  const float* w1   = (const float*)d_in[10];
  const float* b1   = (const float*)d_in[11];
  const float* w2   = (const float*)d_in[12];
  const float* b2   = (const float*)d_in[13];
  const float* lnfg = (const float*)d_in[14];
  const float* lnfb = (const float*)d_in[15];
  const float* lmw  = (const float*)d_in[16];
  const float* lmb  = (const float*)d_in[17];
  float* logits = (float*)d_out;

  char* ws = (char*)d_ws;
  size_t off = 0;
  auto take = [&](size_t bytes) -> void* {
    void* p = ws + off;
    off += (bytes + 255) & ~(size_t)255;
    return p;
  };
  float*  x      = (float*)take((size_t)BT * Dc * 4);
  __bf16* hbuf   = (__bf16*)take((size_t)BT * Dc * 2);
  __bf16* qkvb   = (__bf16*)take((size_t)BT * QKVN * 2);
  __bf16* attnb  = (__bf16*)take((size_t)BT * Dc * 2);
  __bf16* midb   = (__bf16*)take((size_t)BT * FFc * 2);
  __bf16* wt_qkv = (__bf16*)take((size_t)QKVN * Dc * 2);
  __bf16* wt_wo  = (__bf16*)take((size_t)Dc * Dc * 2);
  __bf16* wt_w1  = (__bf16*)take((size_t)FFc * Dc * 2);
  __bf16* wt_w2  = (__bf16*)take((size_t)Dc * FFc * 2);
  __bf16* wt_lm  = (__bf16*)take((size_t)Vc * Dc * 2);

  embed_k<<<BT, 256, 0, stream>>>(idx, tok, pos, x);

  for (int l = 0; l < Lc; ++l) {
    const float* wqkv_l = wqkv + (size_t)l * 3 * Hc * Dc * HDc;
    const float* wo_l = wo + (size_t)l * Dc * Dc;
    const float* w1_l = w1 + (size_t)l * Dc * FFc;
    const float* w2_l = w2 + (size_t)l * FFc * Dc;

    // ln1 -> h
    ln_k<<<BT, 256, 0, stream>>>(x, ln1g + l * Dc, ln1b + l * Dc, hbuf);
    // qkv = h @ Wqkv
    convqkv_k<<<(QKVN * Dc) / 256, 256, 0, stream>>>(wqkv_l, wt_qkv);
    gemm_bf16_k<0><<<dim3(QKVN / 128, BT / 128), 256, 0, stream>>>(
        hbuf, Dc, wt_qkv, Dc, nullptr, nullptr, nullptr, qkvb, QKVN, Dc);
    // attention
    attn_k<<<dim3(Tc / 64, Bc * Hc), 128, 0, stream>>>(qkvb, attnb);
    // x += attn @ Wo + bo
    convT_k<<<(Dc * Dc) / 256, 256, 0, stream>>>(wo_l, wt_wo, Dc, Dc);
    gemm_bf16_k<2><<<dim3(Dc / 128, BT / 128), 256, 0, stream>>>(
        attnb, Dc, wt_wo, Dc, bo + l * Dc, x, x, nullptr, Dc, Dc);
    // ln2 -> h
    ln_k<<<BT, 256, 0, stream>>>(x, ln2g + l * Dc, ln2b + l * Dc, hbuf);
    // mid = gelu(h @ W1 + b1)
    convT_k<<<(Dc * FFc) / 256, 256, 0, stream>>>(w1_l, wt_w1, Dc, FFc);
    gemm_bf16_k<1><<<dim3(FFc / 128, BT / 128), 256, 0, stream>>>(
        hbuf, Dc, wt_w1, Dc, b1 + l * FFc, nullptr, nullptr, midb, FFc, Dc);
    // x += mid @ W2 + b2
    convT_k<<<(FFc * Dc) / 256, 256, 0, stream>>>(w2_l, wt_w2, FFc, Dc);
    gemm_bf16_k<2><<<dim3(Dc / 128, BT / 128), 256, 0, stream>>>(
        midb, FFc, wt_w2, FFc, b2 + l * Dc, x, x, nullptr, Dc, FFc);
  }

  // final LN + LM head
  ln_k<<<BT, 256, 0, stream>>>(x, lnfg, lnfb, hbuf);
  convT_k<<<((size_t)Vc * Dc) / 256, 256, 0, stream>>>(lmw, wt_lm, Dc, Vc);
  gemm_bf16_k<3><<<dim3(Vc / 128, BT / 128), 256, 0, stream>>>(
      hbuf, Dc, wt_lm, Dc, lmb, nullptr, logits, nullptr, Vc, Dc);
}
